// GanglionWithCA_37838661877919
// MI455X (gfx1250) — compile-verified, matrix operand order checked by reading, not run on hardware
//
#include <hip/hip_runtime.h>
#include <hip/hip_bf16.h>

// CA step on 384^3 periodic grid, but only the first 64 flat outputs
// (x=0, y=0, z=0..63) are required. One wave32 does everything:
//   1) colsum[j] = 9-cell (x,y) neighborhood sum at depth z(j), j=0..65,
//      z(j) in {383, 0..64}  (staged in LDS)
//   2) one v_wmma_f32_16x16x32_f16 applies the tridiagonal-band ones
//      matrix => all 64 z-direction box sums (exact: small integers)
//   3) rule lookup (survive/birth bool tables) and store float32.

typedef __attribute__((ext_vector_type(16))) _Float16 v16h;
typedef __attribute__((ext_vector_type(8)))  float    v8f;

#define DIM   384
#define PLANE (DIM * DIM)

__global__ __launch_bounds__(32) void ca_first64_wmma(
    const float* __restrict__ grid,
    const unsigned char* __restrict__ survive,   // 27 x bool (1 byte each)
    const unsigned char* __restrict__ birth,     // 27 x bool (1 byte each)
    float* __restrict__ out)                     // 64 floats
{
    const int lane = threadIdx.x;      // 0..31, single wave
    __shared__ float colsum[66];       // j=0 -> z=383 ; j=1..65 -> z=0..64

    // ---- Step 1: column sums over the 9 periodic (x,y) neighbor rows of (0,0)
    const int xs0 = 383 * DIM * DIM;   // x = -1 (wrap)
    const int xs1 = 0;                 // x =  0
    const int xs2 = 1 * DIM * DIM;     // x = +1
    const int ys0 = 383 * DIM;         // y = -1 (wrap)
    const int ys1 = 0;                 // y =  0
    const int ys2 = 1 * DIM;           // y = +1
    const int xb[3] = {xs0, xs1, xs2};
    const int yb[3] = {ys0, ys1, ys2};

    for (int j = lane; j < 66; j += 32) {
        const int z = (j == 0) ? 383 : (j - 1);
        float s = 0.0f;
        #pragma unroll
        for (int a = 0; a < 3; ++a)
            #pragma unroll
            for (int b = 0; b < 3; ++b)
                s += grid[xb[a] + yb[b] + z];
        colsum[j] = s;
    }
    __syncthreads();

    const int half = lane >> 4;        // 0: lanes 0-15, 1: lanes 16-31
    const int mn   = lane & 15;        // M for A-frag, N for B/D-frags

    // ---- Step 2a: A fragment (16x32 f16):  A[m][k] = colsum[4m+k], k<6
    // 16-bit A layout: lanes 0-15 element e (<8) holds K=e; lanes 16-31 hold
    // K=8..15 in elements 0..7 (all zero here since we only use K<6).
    v16h a = {};
    if (half == 0) {
        #pragma unroll
        for (int k = 0; k < 6; ++k)
            a[k] = (_Float16)colsum[4 * mn + k];
    }

    // ---- Step 2b: B fragment (32x16 f16): band matrix B[k][n] = (n<=k<=n+2, n<4)
    // 16-bit B order used here (contiguous): lanes 0-15 hold K=0..15 (element e
    // <-> K=e), lanes 16-31 hold K=16..31 (all zero here since k<=5).
    v16h b = {};
    #pragma unroll
    for (int e = 0; e < 16; ++e) {
        const int k = 16 * half + e;
        b[e] = (mn < 4 && k >= mn && k <= mn + 2) ? (_Float16)1.0f
                                                  : (_Float16)0.0f;
    }

    // ---- Step 2c: D = A x B  (f32 accum; integer-exact, values <= 27)
    v8f c = {};
    v8f d = __builtin_amdgcn_wmma_f32_16x16x32_f16(
        /*neg_a=*/false, a, /*neg_b=*/false, b,
        /*c_mod=*/(short)0, c, /*reuse_a=*/false, /*reuse_b=*/false);

    // ---- Step 3: rule lookup + store.
    // D layout: lane -> N = lane%16, VGPR v -> M = v + 8*(lane/16).
    // Output z = 4*M + N, valid for N < 4.
    if (mn < 4) {
        #pragma unroll
        for (int v = 0; v < 8; ++v) {
            const int m = v + 8 * half;
            const int z = 4 * m + mn;              // flat index (x=0,y=0,z)
            const float center = grid[z];
            const int ci = (int)(d[v] - center + 0.5f);  // 26-neighbor count
            const bool live = center > 0.5f;
            const unsigned char r = live ? survive[ci] : birth[ci];
            out[z] = r ? 1.0f : 0.0f;
        }
    }
}

extern "C" void kernel_launch(void* const* d_in, const int* in_sizes, int n_in,
                              void* d_out, int out_size, void* d_ws, size_t ws_size,
                              hipStream_t stream) {
    (void)in_sizes; (void)n_in; (void)d_ws; (void)ws_size; (void)out_size;
    const float*         grid    = (const float*)d_in[0];
    const unsigned char* survive = (const unsigned char*)d_in[1]; // jnp bool: 1B/elem
    const unsigned char* birth   = (const unsigned char*)d_in[2];
    // d_in[3] = num_models (== 64, baked into the WMMA mapping)
    float* out = (float*)d_out;

    ca_first64_wmma<<<1, 32, 0, stream>>>(grid, survive, birth, out);
}
